// GatedMultiScaleRetentionLayer_52518860095914
// MI455X (gfx1250) — matrix operand order repaired and checked by reading, not compiled, not run on hardware
//
#include <hip/hip_runtime.h>

// ---------------------------------------------------------------------------
// Types for CDNA5 WMMA (wave32): v_wmma_f32_16x16x32_bf16
// ---------------------------------------------------------------------------
typedef __bf16 bf16_t;
typedef __attribute__((ext_vector_type(16))) __bf16 v16bf;
typedef __attribute__((ext_vector_type(8)))  __bf16 v8bf;
typedef __attribute__((ext_vector_type(8)))  float  v8f;
typedef __attribute__((ext_vector_type(4)))  unsigned int u32x4;
typedef __attribute__((ext_vector_type(8)))  int i32x8;
typedef __attribute__((ext_vector_type(4)))  int i32x4;

// CDNA5 feature probes (fall back to the proven sync path if absent)
#ifndef __has_builtin
#define __has_builtin(x) 0
#endif
#if __has_builtin(__builtin_amdgcn_global_load_async_to_lds_b128)
#define HAVE_ASYNC 1
#else
#define HAVE_ASYNC 0
#endif
#if __has_builtin(__builtin_amdgcn_tensor_load_to_lds) && __has_builtin(__builtin_amdgcn_s_wait_tensorcnt)
#define HAVE_TDM 1
#else
#define HAVE_TDM 0
#endif

#if HAVE_ASYNC
#if __has_builtin(__builtin_amdgcn_s_wait_asynccnt)
#define WAIT_ASYNC0() __builtin_amdgcn_s_wait_asynccnt(0)
#else
#define WAIT_ASYNC0() asm volatile("s_wait_asynccnt 0x0" ::: "memory")
#endif
// builtin signature: (v4i addrspace(1)* gsrc, v4i addrspace(3)* ldst, imm off, imm cpol)
#define GAS(p) ((__attribute__((address_space(1))) i32x4*)(p))
#define LAS(p) ((__attribute__((address_space(3))) i32x4*)(p))
#endif

// A-matrix fragment (16x32 bf16, MxK), per ISA 7.12.2:
//   lanes 0-15 : row M=lane,    elems[0..7]=K(c..c+7),  elems[8..15]=K(c+16..c+23), c=0
//   lanes 16-31: row M=lane-16, same with c=8
// caller passes p = &row[kbase + (lane>>4)*8]
__device__ __forceinline__ v16bf load_frag_a(const bf16_t* p) {
  v8bf lo = *(const v8bf*)p;
  v8bf hi = *(const v8bf*)(p + 16);
  return __builtin_shufflevector(lo, hi, 0,1,2,3,4,5,6,7,8,9,10,11,12,13,14,15);
}

// B-matrix fragment (32x16 bf16, KxN): lane holds column N=lane&15,
// elems[0..7]=K(c..c+7), elems[8..15]=K(c+16..c+23), c=(lane>>4)*8.  caller passes p = &col_row[kbase + (lane>>4)*8]
__device__ __forceinline__ v16bf load_frag_b(const bf16_t* p) {
  v8bf lo = *(const v8bf*)p;
  v8bf hi = *(const v8bf*)(p + 16);
  return __builtin_shufflevector(lo, hi, 0,1,2,3,4,5,6,7,8,9,10,11,12,13,14,15);
}

#define WMMA_BF16(a, b, c) \
  __builtin_amdgcn_wmma_f32_16x16x32_bf16(false, (a), false, (b), (short)0, (c), false, false)

// ---------------------------------------------------------------------------
// fp32 -> bf16 convert (row-major copy)
// ---------------------------------------------------------------------------
__global__ __launch_bounds__(256) void f32_to_bf16_kernel(
    const float* __restrict__ src, bf16_t* __restrict__ dst, int n) {
  int i = (blockIdx.x * 256 + threadIdx.x) * 4;
  if (i + 3 < n) {
    dst[i + 0] = (bf16_t)src[i + 0];
    dst[i + 1] = (bf16_t)src[i + 1];
    dst[i + 2] = (bf16_t)src[i + 2];
    dst[i + 3] = (bf16_t)src[i + 3];
  }
}

// ---------------------------------------------------------------------------
// Transpose fp32 [Kd][N] -> bf16 [N][Kd]  (so GEMM B-frags read contiguous K)
// ---------------------------------------------------------------------------
__global__ __launch_bounds__(256) void transpose_to_bf16_kernel(
    const float* __restrict__ W, bf16_t* __restrict__ Wt, int Kd, int N) {
  __shared__ float tile[32][33];
  int nb = blockIdx.x * 32, kb = blockIdx.y * 32;
  int tx = threadIdx.x & 31, ty = threadIdx.x >> 5;   // 32 x 8
  for (int i = 0; i < 4; i++)
    tile[ty + i * 8][tx] = W[(size_t)(kb + ty + i * 8) * N + nb + tx];
  __syncthreads();
  for (int i = 0; i < 4; i++)
    Wt[(size_t)(nb + ty + i * 8) * Kd + kb + tx] = (bf16_t)tile[tx][ty + i * 8];
}

// ---------------------------------------------------------------------------
// V transpose: vf fp32 [S][H*128] -> vt bf16 [H][128][S]
// ---------------------------------------------------------------------------
__global__ __launch_bounds__(256) void vtrans_kernel(
    const float* __restrict__ vf, bf16_t* __restrict__ vt, int S) {
  __shared__ float tile[32][33];
  int sb = blockIdx.x * 32, db = blockIdx.y * 32, h = blockIdx.z;
  int tx = threadIdx.x & 31, ty = threadIdx.x >> 5;
  for (int i = 0; i < 4; i++)
    tile[ty + i * 8][tx] = vf[(size_t)(sb + ty + i * 8) * 2048 + h * 128 + db + tx];
  __syncthreads();
  for (int i = 0; i < 4; i++)
    vt[((size_t)h * 128 + db + ty + i * 8) * S + sb + tx] = (bf16_t)tile[tx][ty + i * 8];
}

// ---------------------------------------------------------------------------
// GEMM: C[M][N] = A[M][Kd] @ Bt[N][Kd]^T + bias, optional SiLU.
// bf16 inputs, f32 accumulate via v_wmma_f32_16x16x32_bf16.
// 128x128 tile / workgroup, 8 waves, each wave 32x64 (2x4 WMMA accs).
// Async path: double-buffered LDS fed by global_load_async_to_lds_b128,
// one barrier per K-step, next tile streams in during WMMA.
// ---------------------------------------------------------------------------
#define GLSA 48   // LDS stride (elems) for 32-wide K panels; 96B, 16B aligned

__global__ __launch_bounds__(256) void gemm_bf16_kernel(
    const bf16_t* __restrict__ A, const bf16_t* __restrict__ Bt,
    const float* __restrict__ bias, float* __restrict__ C,
    int M, int N, int Kd, int act) {
  const int tid  = threadIdx.x;
  const int lane = tid & 31, w = tid >> 5;
  const int l15 = lane & 15, lhi = lane >> 4;
  const int m0 = blockIdx.y * 128, n0 = blockIdx.x * 128;
  const int wm = (w & 3) * 32, wn = (w >> 2) * 64;
  const int lr = tid >> 1;              // tile row 0..127
  const int lc = (tid & 1) * 16;        // tile col 0 or 16 (x16 bf16 = 32B)

  v8f acc[2][4] = {};

#if HAVE_ASYNC
  __shared__ bf16_t As[2][128 * GLSA];
  __shared__ bf16_t Bs[2][128 * GLSA];
  auto issue = [&](int buf, int k0) {
    const bf16_t* ga = A  + (size_t)(m0 + lr) * Kd + k0 + lc;
    const bf16_t* gb = Bt + (size_t)(n0 + lr) * Kd + k0 + lc;
    bf16_t* la = &As[buf][lr * GLSA + lc];
    bf16_t* lb = &Bs[buf][lr * GLSA + lc];
    __builtin_amdgcn_global_load_async_to_lds_b128(GAS(ga),     LAS(la),     0, 0);
    __builtin_amdgcn_global_load_async_to_lds_b128(GAS(ga + 8), LAS(la + 8), 0, 0);
    __builtin_amdgcn_global_load_async_to_lds_b128(GAS(gb),     LAS(lb),     0, 0);
    __builtin_amdgcn_global_load_async_to_lds_b128(GAS(gb + 8), LAS(lb + 8), 0, 0);
  };
  issue(0, 0);
  const int nIt = Kd >> 5;
  for (int it = 0; it < nIt; ++it) {
    const int cur = it & 1;
    WAIT_ASYNC0();                       // our async writes for buf `cur` landed
    __syncthreads();                     // everyone's landed; prev reads done
    if (it + 1 < nIt) issue(cur ^ 1, (it + 1) << 5);

    v16bf af[2], bfr[4];
    for (int mf = 0; mf < 2; mf++)
      af[mf] = load_frag_a(&As[cur][(wm + mf * 16 + l15) * GLSA + lhi * 8]);
    for (int nf = 0; nf < 4; nf++)
      bfr[nf] = load_frag_b(&Bs[cur][(wn + nf * 16 + l15) * GLSA + lhi * 8]);
    for (int mf = 0; mf < 2; mf++)
      for (int nf = 0; nf < 4; nf++)
        acc[mf][nf] = WMMA_BF16(af[mf], bfr[nf], acc[mf][nf]);
  }
#else
  __shared__ bf16_t As[128 * GLSA];
  __shared__ bf16_t Bs[128 * GLSA];
  for (int k0 = 0; k0 < Kd; k0 += 32) {
    const uint4* ga = (const uint4*)(A  + (size_t)(m0 + lr) * Kd + k0 + lc);
    const uint4* gb = (const uint4*)(Bt + (size_t)(n0 + lr) * Kd + k0 + lc);
    uint4 a0 = ga[0], a1 = ga[1];
    uint4 b0 = gb[0], b1 = gb[1];
    __syncthreads();
    *(uint4*)(As + lr * GLSA + lc)     = a0;
    *(uint4*)(As + lr * GLSA + lc + 8) = a1;
    *(uint4*)(Bs + lr * GLSA + lc)     = b0;
    *(uint4*)(Bs + lr * GLSA + lc + 8) = b1;
    __syncthreads();
    v16bf af[2], bfr[4];
    for (int mf = 0; mf < 2; mf++)
      af[mf] = load_frag_a(As + (wm + mf * 16 + l15) * GLSA + lhi * 8);
    for (int nf = 0; nf < 4; nf++)
      bfr[nf] = load_frag_b(Bs + (wn + nf * 16 + l15) * GLSA + lhi * 8);
    for (int mf = 0; mf < 2; mf++)
      for (int nf = 0; nf < 4; nf++)
        acc[mf][nf] = WMMA_BF16(af[mf], bfr[nf], acc[mf][nf]);
  }
#endif

  // epilogue: C layout per lane/vgpr: m = r + 8*lhi, n = l15
  for (int mf = 0; mf < 2; mf++)
    for (int nf = 0; nf < 4; nf++) {
      int n = n0 + wn + nf * 16 + l15;
      float bv = bias[n];
      for (int r = 0; r < 8; r++) {
        int m = m0 + wm + mf * 16 + r + 8 * lhi;
        float v = acc[mf][nf][r] + bv;
        if (act == 1) v = v / (1.f + __expf(-v));   // SiLU
        C[(size_t)m * N + n] = v;
      }
    }
}

// ---------------------------------------------------------------------------
// RoPE + xPos + decay pre-scaling.
// qf/kf fp32 [S][H*128] -> qp/kp bf16 [H][S][128]
//   q' = rope(q)*scale*gamma^s,  k' = rope(k)/scale*gamma^-s
// ---------------------------------------------------------------------------
__global__ __launch_bounds__(256) void rope_kernel(
    const float* __restrict__ qf, const float* __restrict__ kf,
    bf16_t* __restrict__ qp, bf16_t* __restrict__ kp, int S) {
  int idx = blockIdx.x * 256 + threadIdx.x;   // (s, h, pair)
  int p = idx & 63;
  int h = (idx >> 6) & 15;
  int s = idx >> 10;
  int d0 = 2 * p;
  size_t src = (size_t)s * 2048 + h * 128 + d0;
  float q0 = qf[src], q1 = qf[src + 1];
  float k0 = kf[src], k1 = kf[src + 1];

  float inv_freq = __powf(10000.f, -((float)d0) / 128.f);
  float ang = (float)s * inv_freq;
  float sn, c;
  __sincosf(ang, &sn, &c);

  float sbase = ((float)d0 + 0.4f * 128.f) / (1.4f * 128.f);
  float power = ((float)s - (float)(S / 2)) / 512.f;
  float scale = __powf(sbase, power);

  float gamma = 1.f - exp2f(-5.f - (float)h);
  float lg2 = log2f(gamma);
  float gq = exp2f((float)s * lg2);    // gamma^s
  float gk = exp2f(-(float)s * lg2);   // gamma^-s

  float rq0 = (q0 * c - q1 * sn) * scale * gq;
  float rq1 = (q1 * c + q0 * sn) * scale * gq;
  float inv_scale = 1.f / scale;
  float rk0 = (k0 * c - k1 * sn) * inv_scale * gk;
  float rk1 = (k1 * c + k0 * sn) * inv_scale * gk;

  size_t dst = ((size_t)h * S + s) * 128 + d0;
  qp[dst]     = (bf16_t)rq0;
  qp[dst + 1] = (bf16_t)rq1;
  kp[dst]     = (bf16_t)rk0;
  kp[dst + 1] = (bf16_t)rk1;
}

// ---------------------------------------------------------------------------
// Retention core (per head, flash-style over 64-row query tiles):
//   scores = q' @ k'^T  (decay folded into q',k'), causal mask on diagonal tile
//   ret    = scores @ v
// Then GroupNorm over V_DIM=128 per row and SiLU-gate multiply -> gated bf16.
// Async path: Q preloaded via TDM tensor_load_to_lds (DMA pads rows to the
// 272B LDS stride), K/V tiles double-buffered via async-to-LDS loads.
// ---------------------------------------------------------------------------
#define QSTR 136   // Qs/Ks LDS stride (elems) : 272B, 16B aligned
#define VSTR 72    // Vts/Ss LDS stride (elems): 144B, 16B aligned
#define RSTR 132   // retS stride (floats)     : 528B, 16B aligned

#if HAVE_ASYNC
#define RET_SMEM_BYTES (3 * (64 * QSTR) * 2 + 2 * (128 * VSTR) * 2 + (64 * VSTR) * 2)
#else
#define RET_SMEM_BYTES (2 * (64 * QSTR) * 2 + (128 * VSTR) * 2 + (64 * VSTR) * 2)
#endif

__global__ __launch_bounds__(256) void retention_kernel(
    const bf16_t* __restrict__ qp,  // [H][S][128]
    const bf16_t* __restrict__ kp,  // [H][S][128]
    const bf16_t* __restrict__ vt,  // [H][128][S]
    const float*  __restrict__ gf,  // [S][H*128] (silu already applied)
    bf16_t* __restrict__ gated,     // [S][H*128]
    int S) {
  __shared__ __attribute__((aligned(16))) char smem[RET_SMEM_BYTES];
  bf16_t* Qs = (bf16_t*)smem;             // 64 x QSTR
#if HAVE_ASYNC
  bf16_t* KsB[2] = { Qs + 64 * QSTR, Qs + 2 * 64 * QSTR };
  bf16_t* VtB[2] = { Qs + 3 * 64 * QSTR, Qs + 3 * 64 * QSTR + 128 * VSTR };
  bf16_t* Ss = Qs + 3 * 64 * QSTR + 2 * 128 * VSTR;
#else
  bf16_t* KsB[2] = { Qs + 64 * QSTR, Qs + 64 * QSTR };
  bf16_t* VtB[2] = { Qs + 2 * 64 * QSTR, Qs + 2 * 64 * QSTR };
  bf16_t* Ss = Qs + 2 * 64 * QSTR + 128 * VSTR;
#endif
  float* retS = (float*)smem;             // overlay on Qs+Ks0 after main loop
  __shared__ float muA[64], invA[64];

  const int I = blockIdx.x, h = blockIdx.y;
  const int tid = threadIdx.x, lane = tid & 31, w = tid >> 5;
  const int l15 = lane & 15, lhi = lane >> 4;
  const int wm = w & 3, nh = w >> 2;
  const size_t hbase = (size_t)h * S * 128;

#if HAVE_ASYNC
#if HAVE_TDM
  // Q tile preload via Tensor Data Mover: 2D tile 64x128 bf16, DMA inserts
  // 16B of padding every 256B so LDS rows land at the QSTR(=272B) stride.
  if (w == 0) {
    unsigned lds_addr = (unsigned)(unsigned long long)
        (__attribute__((address_space(3))) char*)(void*)Qs;
    unsigned long long gaddr =
        (unsigned long long)(qp + hbase + (size_t)(I * 64) * 128);
    u32x4 g0;
    g0[0] = 1u;                                           // count=1 (valid D#)
    g0[1] = lds_addr;                                     // lds_addr
    g0[2] = (unsigned)(gaddr & 0xFFFFFFFFu);              // global_addr lo
    g0[3] = (unsigned)((gaddr >> 32) & 0x1FFFFFFu) | (2u << 30); // hi | type=2
    i32x8 g1;
    g1[0] = (1 << 16) | (1 << 20) | (5 << 22) | (3 << 25); // 2B elems, pad on,
                                                           // every 64 DW pad 4 DW
    g1[1] = (int)(128u << 16);   // tensor_dim0 = 128 (bits 79:48)
    g1[2] = (int)(2048u << 16);  // tensor_dim1 = 2048 (bits 111:80)
    g1[3] = (int)(128u << 16);   // tile_dim0 = 128 (bits 127:112)
    g1[4] = 64;                  // tile_dim1 = 64
    g1[5] = 128;                 // tensor_dim0_stride = 128
    g1[6] = 0;
    g1[7] = 0;
    i32x4 gz4 = {0, 0, 0, 0};
    i32x8 gz8 = {0, 0, 0, 0, 0, 0, 0, 0};
    __builtin_amdgcn_tensor_load_to_lds(g0, g1, gz4, gz4, gz8, 0);
  }
#else
  {
    int r = tid >> 2, c0 = (tid & 3) * 32;
    const bf16_t* g = qp + hbase + (size_t)(I * 64 + r) * 128 + c0;
    bf16_t* d = Qs + r * QSTR + c0;
    __builtin_amdgcn_global_load_async_to_lds_b128(GAS(g),      LAS(d),      0, 0);
    __builtin_amdgcn_global_load_async_to_lds_b128(GAS(g + 8),  LAS(d + 8),  0, 0);
    __builtin_amdgcn_global_load_async_to_lds_b128(GAS(g + 16), LAS(d + 16), 0, 0);
    __builtin_amdgcn_global_load_async_to_lds_b128(GAS(g + 24), LAS(d + 24), 0, 0);
  }
#endif
  auto issueKV = [&](int buf, int J) {
    int r = tid >> 2, c0 = (tid & 3) * 32;
    const bf16_t* gk = kp + hbase + (size_t)(J * 64 + r) * 128 + c0;
    bf16_t* dk = KsB[buf] + r * QSTR + c0;
    __builtin_amdgcn_global_load_async_to_lds_b128(GAS(gk),      LAS(dk),      0, 0);
    __builtin_amdgcn_global_load_async_to_lds_b128(GAS(gk + 8),  LAS(dk + 8),  0, 0);
    __builtin_amdgcn_global_load_async_to_lds_b128(GAS(gk + 16), LAS(dk + 16), 0, 0);
    __builtin_amdgcn_global_load_async_to_lds_b128(GAS(gk + 24), LAS(dk + 24), 0, 0);
    int dvr = tid >> 1, e0 = (tid & 1) * 32;
    const bf16_t* gv = vt + hbase + (size_t)dvr * S + J * 64 + e0;
    bf16_t* dv = VtB[buf] + dvr * VSTR + e0;
    __builtin_amdgcn_global_load_async_to_lds_b128(GAS(gv),      LAS(dv),      0, 0);
    __builtin_amdgcn_global_load_async_to_lds_b128(GAS(gv + 8),  LAS(dv + 8),  0, 0);
    __builtin_amdgcn_global_load_async_to_lds_b128(GAS(gv + 16), LAS(dv + 16), 0, 0);
    __builtin_amdgcn_global_load_async_to_lds_b128(GAS(gv + 24), LAS(dv + 24), 0, 0);
  };
  issueKV(0, 0);
#else
  {  // Q tile preload (register-staged)
    int r = tid >> 2, c0 = (tid & 3) * 32;
    const uint4* g = (const uint4*)(qp + hbase + (size_t)(I * 64 + r) * 128 + c0);
    uint4 x0 = g[0], x1 = g[1], x2 = g[2], x3 = g[3];
    uint4* d = (uint4*)(Qs + r * QSTR + c0);
    d[0] = x0; d[1] = x1; d[2] = x2; d[3] = x3;
  }
#endif

  v8f acc2[4] = {};

  for (int J = 0; J <= I; J++) {
    const int cur = J & 1;
#if HAVE_ASYNC
    WAIT_ASYNC0();
#if HAVE_TDM
    __builtin_amdgcn_s_wait_tensorcnt(0);
#endif
    __syncthreads();                  // tiles for `cur` visible; prev reads done
    if (J < I) issueKV(cur ^ 1, J + 1);
#else
    __syncthreads();
    {
      int r = tid >> 2, c0 = (tid & 3) * 32;
      const uint4* gk = (const uint4*)(kp + hbase + (size_t)(J * 64 + r) * 128 + c0);
      uint4 k0 = gk[0], k1 = gk[1], k2 = gk[2], k3 = gk[3];
      uint4* dk = (uint4*)(KsB[0] + r * QSTR + c0);
      dk[0] = k0; dk[1] = k1; dk[2] = k2; dk[3] = k3;
      int dvr = tid >> 1, e0 = (tid & 1) * 32;
      const uint4* gv = (const uint4*)(vt + hbase + (size_t)dvr * S + J * 64 + e0);
      uint4 v0 = gv[0], v1 = gv[1], v2 = gv[2], v3 = gv[3];
      uint4* dv = (uint4*)(VtB[0] + dvr * VSTR + e0);
      dv[0] = v0; dv[1] = v1; dv[2] = v2; dv[3] = v3;
    }
    __syncthreads();
#endif
    const bf16_t* Kcur = KsB[cur];
    const bf16_t* Vcur = VtB[cur];

    // ---- stage 1: scores (64x64) = Q(64x128) @ K(64x128)^T, 16 tiles of 16x16
    for (int tt = 0; tt < 2; tt++) {
      int t = w * 2 + tt, tm = t >> 2, tn = t & 3;
      v8f sacc = {};
      for (int kk = 0; kk < 128; kk += 32) {
        v16bf aq = load_frag_a(Qs   + (tm * 16 + l15) * QSTR + kk + lhi * 8);
        v16bf bk = load_frag_b(Kcur + (tn * 16 + l15) * QSTR + kk + lhi * 8);
        sacc = WMMA_BF16(aq, bk, sacc);
      }
      int colg = tn * 16 + l15;
      for (int r = 0; r < 8; r++) {
        int rowg = tm * 16 + r + 8 * lhi;
        float sv = sacc[r];
        if (J == I && colg > rowg) sv = 0.f;   // causal mask on diagonal tile
        Ss[rowg * VSTR + colg] = (bf16_t)sv;
      }
    }
    __syncthreads();

    // ---- stage 2: ret(64x128) += scores(64x64) @ V(64x128); Vt gives B-frags
    for (int kk2 = 0; kk2 < 64; kk2 += 32) {
      v16bf as = load_frag_a(Ss + (wm * 16 + l15) * VSTR + kk2 + lhi * 8);
      for (int nf = 0; nf < 4; nf++) {
        v16bf bv = load_frag_b(Vcur + (nh * 64 + nf * 16 + l15) * VSTR + kk2 + lhi * 8);
        acc2[nf] = WMMA_BF16(as, bv, acc2[nf]);
      }
    }
  }

  // ---- epilogue: GroupNorm over 128 dims per row, multiply by silu gate
  __syncthreads();   // Qs/Ks no longer needed; retS overlays them
  for (int nf = 0; nf < 4; nf++)
    for (int r = 0; r < 8; r++) {
      int row = wm * 16 + r + 8 * lhi;
      int col = nh * 64 + nf * 16 + l15;
      retS[row * RSTR + col] = acc2[nf][r];
    }
  __syncthreads();
  if (tid < 64) {
    float s1 = 0.f, s2 = 0.f;
    for (int d2 = 0; d2 < 128; d2++) {
      float v = retS[tid * RSTR + d2];
      s1 += v; s2 += v * v;
    }
    float mu = s1 * (1.f / 128.f);
    float var = s2 * (1.f / 128.f) - mu * mu;
    muA[tid] = mu;
    invA[tid] = rsqrtf(var + 1e-5f);
  }
  __syncthreads();
  {
    int row = tid >> 2, c0 = (tid & 3) * 32;
    int gs = I * 64 + row;
    float mu = muA[row], inv = invA[row];
    for (int d2 = 0; d2 < 32; d2++) {
      int dd = c0 + d2;
      float v = (retS[row * RSTR + dd] - mu) * inv;
      float g = gf[(size_t)gs * 2048 + h * 128 + dd];
      gated[(size_t)gs * 2048 + h * 128 + dd] = (bf16_t)(v * g);
    }
  }
}

// ---------------------------------------------------------------------------
// Launch
// ---------------------------------------------------------------------------
extern "C" void kernel_launch(void* const* d_in, const int* in_sizes, int n_in,
                              void* d_out, int out_size, void* d_ws, size_t ws_size,
                              hipStream_t stream) {
  const float* x  = (const float*)d_in[0];
  const float* Wq = (const float*)d_in[1]; const float* bq = (const float*)d_in[2];
  const float* Wk = (const float*)d_in[3]; const float* bk = (const float*)d_in[4];
  const float* Wv = (const float*)d_in[5]; const float* bv = (const float*)d_in[6];
  const float* Wg = (const float*)d_in[7]; const float* bg = (const float*)d_in[8];
  const float* Wo = (const float*)d_in[9]; const float* bo = (const float*)d_in[10];

  const int S = 2048, D = 2048, H = 16;
  char* ws = (char*)d_ws;
  size_t off = 0;
  auto alloc = [&](size_t bytes) {
    size_t o = off;
    off += (bytes + 255) & ~(size_t)255;
    return o;
  };
  bf16_t* xb    = (bf16_t*)(ws + alloc((size_t)S * D * 2));
  bf16_t* Wt    = (bf16_t*)(ws + alloc((size_t)D * D * 2));   // reused for all 5 weights
  float*  qf    = (float*) (ws + alloc((size_t)S * D * 4));
  float*  kf    = (float*) (ws + alloc((size_t)S * D * 4));
  float*  vf    = (float*) (ws + alloc((size_t)S * D * 4));
  float*  gfbuf = (float*) (ws + alloc((size_t)S * D * 4));
  bf16_t* qp    = (bf16_t*)(ws + alloc((size_t)H * S * 128 * 2));
  bf16_t* kp    = (bf16_t*)(ws + alloc((size_t)H * S * 128 * 2));
  bf16_t* vt    = (bf16_t*)(ws + alloc((size_t)H * 128 * S * 2));
  bf16_t* gated = (bf16_t*)(ws + alloc((size_t)S * D * 2));
  (void)ws_size; (void)in_sizes; (void)n_in; (void)out_size;

  dim3 b256(256);
  dim3 tgrid(D / 32, D / 32);
  dim3 ggrid(D / 128, S / 128);

  f32_to_bf16_kernel<<<(S * D) / 1024, b256, 0, stream>>>(x, xb, S * D);

  transpose_to_bf16_kernel<<<tgrid, b256, 0, stream>>>(Wq, Wt, D, D);
  gemm_bf16_kernel<<<ggrid, b256, 0, stream>>>(xb, Wt, bq, qf, S, D, D, 0);

  transpose_to_bf16_kernel<<<tgrid, b256, 0, stream>>>(Wk, Wt, D, D);
  gemm_bf16_kernel<<<ggrid, b256, 0, stream>>>(xb, Wt, bk, kf, S, D, D, 0);

  transpose_to_bf16_kernel<<<tgrid, b256, 0, stream>>>(Wv, Wt, D, D);
  gemm_bf16_kernel<<<ggrid, b256, 0, stream>>>(xb, Wt, bv, vf, S, D, D, 0);

  transpose_to_bf16_kernel<<<tgrid, b256, 0, stream>>>(Wg, Wt, D, D);
  gemm_bf16_kernel<<<ggrid, b256, 0, stream>>>(xb, Wt, bg, gfbuf, S, D, D, 1);  // SiLU

  rope_kernel<<<(S * H * 64) / 256, b256, 0, stream>>>(qf, kf, qp, kp, S);
  vtrans_kernel<<<dim3(S / 32, 4, H), b256, 0, stream>>>(vf, vt, S);

  retention_kernel<<<dim3(S / 64, H), b256, 0, stream>>>(qp, kp, vt, gfbuf, gated, S);

  transpose_to_bf16_kernel<<<tgrid, b256, 0, stream>>>(Wo, Wt, D, D);
  gemm_bf16_kernel<<<ggrid, b256, 0, stream>>>(gated, Wt, bo, (float*)d_out, S, D, D, 0);
}
